// DBTransformer_90383291777371
// MI455X (gfx1250) — compile-verified, matrix operand order checked by reading, not run on hardware
//
#include <hip/hip_runtime.h>
#include <stdint.h>

// ---------------------------------------------------------------------------
// Model constants (from reference)
// ---------------------------------------------------------------------------
#define DIM        256
#define HEADS      8
#define DIM_HEAD   16
#define INNER      128
#define FF         1024
#define DIM_OUT    16
#define TOK_SLOTS  16            // padded sequence capacity (max real s = 14)
#define TOK_STRIDE (TOK_SLOTS * DIM)  // 4096 halves per row
#define RPB        2             // rows (M-tiles) per workgroup: B-frag reuse

typedef __attribute__((ext_vector_type(16))) _Float16 v16h;
typedef __attribute__((ext_vector_type(8)))  _Float16 v8h;
typedef __attribute__((ext_vector_type(8)))  float    v8f;

union AFrag { v16h v; v8h p[2]; };

static __device__ __forceinline__ v8f v8f_zero() {
  v8f c;
#pragma unroll
  for (int i = 0; i < 8; ++i) c[i] = 0.f;
  return c;
}

// ---------------------------------------------------------------------------
// Two 16x16 output tiles (two M-tiles = two table rows) sharing one B stream,
// accumulated over K with v_wmma_f32_16x16x32_f16.  Each B fragment (two
// contiguous 16B global_load_b128 per lane from the transposed f16 weight)
// feeds TWO WMMAs -> halves L2 weight traffic vs one-row-per-block.
// Operand layout (ISA 7.12.2, 16-bit): lane L -> row/col L&15; half i ->
// K = i + (i>=8 ? 8 : 0) + (L>>4)*8  => two contiguous 8-half (16B) runs.
// ---------------------------------------------------------------------------
template <int K>
static __device__ __forceinline__ void wmma_tile2(
    const _Float16* A0, const _Float16* A1, int lda,
    const _Float16* Bt, int ldb, int lane, v8f& c0, v8f& c1) {
  const int g = (lane >> 4) & 1;
  const int r = lane & 15;
  const _Float16* a0p = A0 + r * lda + g * 8;
  const _Float16* a1p = A1 + r * lda + g * 8;
  const _Float16* bp  = Bt + r * ldb + g * 8;
#pragma unroll
  for (int k0 = 0; k0 < K; k0 += 32) {
    AFrag a0, a1, b;
    b.p[0]  = *(const v8h*)(bp + k0);
    b.p[1]  = *(const v8h*)(bp + k0 + 16);
    a0.p[0] = *(const v8h*)(a0p + k0);
    a0.p[1] = *(const v8h*)(a0p + k0 + 16);
    a1.p[0] = *(const v8h*)(a1p + k0);
    a1.p[1] = *(const v8h*)(a1p + k0 + 16);
    c0 = __builtin_amdgcn_wmma_f32_16x16x32_f16(false, a0.v, false, b.v,
                                                (short)0, c0, false, false);
    c1 = __builtin_amdgcn_wmma_f32_16x16x32_f16(false, a1.v, false, b.v,
                                                (short)0, c1, false, false);
  }
}

// Single-tile version (final head kernel).
template <int K>
static __device__ __forceinline__ v8f wmma_tile_acc(const _Float16* A, int lda,
                                                    const _Float16* Bt, int ldb,
                                                    int lane, v8f c) {
  const int g = (lane >> 4) & 1;
  const int r = lane & 15;
  const _Float16* ap = A  + r * lda + g * 8;
  const _Float16* bp = Bt + r * ldb + g * 8;
#pragma unroll
  for (int k0 = 0; k0 < K; k0 += 32) {
    AFrag a, b;
    a.p[0] = *(const v8h*)(ap + k0);
    a.p[1] = *(const v8h*)(ap + k0 + 16);
    b.p[0] = *(const v8h*)(bp + k0);
    b.p[1] = *(const v8h*)(bp + k0 + 16);
    c = __builtin_amdgcn_wmma_f32_16x16x32_f16(false, a.v, false, b.v,
                                               (short)0, c, false, false);
  }
  return c;
}

static __device__ __forceinline__ float gelu_tanh(float x) {
  float x3 = x * x * x;
  return 0.5f * x * (1.f + tanhf(0.7978845608028654f * (x + 0.044715f * x3)));
}

// LayerNorm of 16 rows x 256 cols (f32 LDS) into f16 LDS (stride 264).
// 128 threads: 8 threads per row, 32 cols each; partial sums via LDS.
static __device__ __forceinline__ void ln_rows(const float* Xs, _Float16* Hs,
                                               const float* gg, const float* bb,
                                               float* red, int tid,
                                               bool relu_after) {
  const int row = tid >> 3, sub = tid & 7;
  const float* xr = Xs + row * 256 + sub * 32;
  float s0 = 0.f, s1 = 0.f;
#pragma unroll
  for (int i = 0; i < 32; ++i) { float v = xr[i]; s0 += v; s1 += v * v; }
  red[(row * 8 + sub) * 2 + 0] = s0;
  red[(row * 8 + sub) * 2 + 1] = s1;
  __syncthreads();
  float m = 0.f, q = 0.f;
#pragma unroll
  for (int i = 0; i < 8; ++i) {
    m += red[(row * 8 + i) * 2 + 0];
    q += red[(row * 8 + i) * 2 + 1];
  }
  m *= (1.f / 256.f);
  q = q * (1.f / 256.f) - m * m;
  float rs = rsqrtf(q + 1e-5f);
#pragma unroll
  for (int i = 0; i < 32; ++i) {
    int cc = sub * 32 + i;
    float v = (xr[i] - m) * rs * gg[cc] + bb[cc];
    if (relu_after) v = fmaxf(v, 0.f);
    Hs[row * 264 + cc] = (_Float16)v;
  }
}

// ---------------------------------------------------------------------------
// Weight transpose + f32 -> f16:  dst[n*K + k] = (f16) src[k*N + n]
// ---------------------------------------------------------------------------
__global__ __launch_bounds__(256) void xpose_f32_to_f16(
    const float* __restrict__ src, _Float16* __restrict__ dst, int K, int N) {
  int idx = blockIdx.x * 256 + threadIdx.x;
  if (idx < K * N) {
    int k = idx / N, n = idx - k * N;
    dst[n * K + k] = (_Float16)src[idx];
  }
}

// ---------------------------------------------------------------------------
// Embedding: slot 0 (CLS) zeroed (filled by layer kernel from param);
// slots 1..ncat = gathered cat embeddings; next nnum slots = num*nw+nb;
// remaining slots zero.  Offsets: NUM_SPECIAL=2, all cats=12 -> off_j = 2+12j.
// ---------------------------------------------------------------------------
__global__ __launch_bounds__(256) void embed_kernel(
    const int* __restrict__ categ, const float* __restrict__ numer,
    const float* __restrict__ emb, const float* __restrict__ nw,
    const float* __restrict__ nb, _Float16* __restrict__ tok,
    int ncat, int nnum) {
  const int row = blockIdx.x, d = threadIdx.x;
  const size_t base = (size_t)row * TOK_STRIDE;
  tok[base + d] = (_Float16)0.f;
  for (int j = 0; j < ncat; ++j) {
    int v = categ[row * ncat + j] + (2 + 12 * j);
    tok[base + (size_t)(1 + j) * DIM + d] = (_Float16)emb[(size_t)v * DIM + d];
  }
  for (int j = 0; j < nnum; ++j) {
    float x = numer[row * nnum + j];
    tok[base + (size_t)(1 + ncat + j) * DIM + d] =
        (_Float16)(x * nw[j * DIM + d] + nb[j * DIM + d]);
  }
  for (int sl = 1 + ncat + nnum; sl < TOK_SLOTS; ++sl)
    tok[base + (size_t)sl * DIM + d] = (_Float16)0.f;
}

// ---------------------------------------------------------------------------
// Full transformer block, RPB=2 table rows per workgroup (4 wave32).
// Sequence padded to 16 rows; pad rows stay row-local garbage and are never
// emitted (tok writes guarded by m < s, cls uses row 0 only).
// ---------------------------------------------------------------------------
struct LayerW {
  const _Float16 *WqkvT, *WoT, *Wf1T, *Wf2T, *clsWT, *tlWT;  // f16 [N][K]
  const float *clsTok;                                       // [256]
  const float *ln1g, *ln1b, *ln2g, *ln2b;
  const float *bf1, *bf2, *clsB, *tlg, *tlb, *tlB;
};

__global__ __launch_bounds__(128) void table_block_kernel(
    const _Float16* __restrict__ tokIn, _Float16* __restrict__ tokOut,
    float* __restrict__ clsOut, LayerW W, int s) {
  const int tid  = threadIdx.x;
  const int wave = tid >> 5, lane = tid & 31;
  const int g = lane >> 4, r = lane & 15;
  const int row0 = blockIdx.x * RPB;

  __shared__ float    Xs[RPB][16 * 256];    // residual streams (f32)
  __shared__ _Float16 Hs[RPB][16 * 264];    // A operands (f16, padded stride)
  __shared__ _Float16 Fs[RPB][16 * 1040];   // FFN acts (f16) / QKV (f32 view)
  __shared__ float    red[16 * 8 * 2];

  // 1) residual load: row 0 = learned CLS, rows 1..s-1 = input tokens, pad 0
  for (int idx = tid; idx < RPB * 16 * 256; idx += 128) {
    int rr = idx >> 12, li = idx & 4095;
    int m = li >> 8;
    float v;
    if (m == 0)      v = W.clsTok[li & 255];
    else if (m < s)  v = (float)tokIn[(size_t)(row0 + rr) * TOK_STRIDE + li];
    else             v = 0.f;
    Xs[rr][li] = v;
  }
  __syncthreads();

  // 2) ln1 -> Hs (both rows)
  ln_rows(Xs[0], Hs[0], W.ln1g, W.ln1b, red, tid, false);
  __syncthreads();
  ln_rows(Xs[1], Hs[1], W.ln1g, W.ln1b, red, tid, false);
  __syncthreads();

  // 3) qkv = ln1(x) @ Wqkv   [16x256]@[256x384] -> QKV f32 in LDS (Fs alias)
  {
    float* QKV0 = reinterpret_cast<float*>(Fs[0]);  // [16][520]
    float* QKV1 = reinterpret_cast<float*>(Fs[1]);
    for (int nt = wave; nt < 24; nt += 4) {
      v8f c0 = v8f_zero(), c1 = v8f_zero();
      wmma_tile2<256>(Hs[0], Hs[1], 264, W.WqkvT + nt * 16 * 256, 256, lane,
                      c0, c1);
#pragma unroll
      for (int j = 0; j < 8; ++j) {
        int o = (j + 8 * g) * 520 + nt * 16 + r;
        QKV0[o] = c0[j];
        QKV1[o] = c1[j];
      }
    }
  }
  __syncthreads();

  // 4) attention: 128 lanes = 8 heads x 16 queries; one row per pass
  for (int rr = 0; rr < RPB; ++rr) {
    const float* QKV = reinterpret_cast<const float*>(Fs[rr]);
    const int head = tid >> 4, m = tid & 15;
    float sc[16];
    float mx = -1e30f;
#pragma unroll
    for (int k = 0; k < 16; ++k) {
      float d = -1e30f;
      if (k < s) {
        d = 0.f;
#pragma unroll
        for (int e = 0; e < 16; ++e)
          d += QKV[m * 520 + head * 16 + e] *
               QKV[k * 520 + INNER + head * 16 + e];
        d *= 0.25f;  // DIM_HEAD^-0.5
      }
      sc[k] = d;
      mx = fmaxf(mx, d);
    }
    float sum = 0.f;
#pragma unroll
    for (int k = 0; k < 16; ++k) { float e = __expf(sc[k] - mx); sc[k] = e; sum += e; }
    float inv = 1.f / sum;
#pragma unroll
    for (int e = 0; e < 16; ++e) {
      float o = 0.f;
#pragma unroll
      for (int k = 0; k < 16; ++k)
        o += sc[k] * QKV[k * 520 + 2 * INNER + head * 16 + e];
      Hs[rr][m * 264 + head * 16 + e] = (_Float16)(o * inv);  // o -> A operand
    }
  }
  __syncthreads();

  // 5) x += o @ Wo   [16x128]@[128x256]
  for (int nt = wave; nt < 16; nt += 4) {
    v8f c0 = v8f_zero(), c1 = v8f_zero();
    wmma_tile2<128>(Hs[0], Hs[1], 264, W.WoT + nt * 16 * 128, 128, lane,
                    c0, c1);
#pragma unroll
    for (int j = 0; j < 8; ++j) {
      int o = (j + 8 * g) * 256 + nt * 16 + r;
      Xs[0][o] += c0[j];
      Xs[1][o] += c1[j];
    }
  }
  __syncthreads();

  // 6) ln2 -> Hs
  ln_rows(Xs[0], Hs[0], W.ln2g, W.ln2b, red, tid, false);
  __syncthreads();
  ln_rows(Xs[1], Hs[1], W.ln2g, W.ln2b, red, tid, false);
  __syncthreads();

  // 7) F = gelu(ln2(x) @ Wf1 + bf1)  [16x256]@[256x1024] -> f16 LDS
  for (int nt = wave; nt < 64; nt += 4) {
    v8f c0 = v8f_zero(), c1 = v8f_zero();
    wmma_tile2<256>(Hs[0], Hs[1], 264, W.Wf1T + nt * 16 * 256, 256, lane,
                    c0, c1);
#pragma unroll
    for (int j = 0; j < 8; ++j) {
      int n = nt * 16 + r;
      int o = (j + 8 * g) * 1040 + n;
      float b1 = W.bf1[n];
      Fs[0][o] = (_Float16)gelu_tanh(c0[j] + b1);
      Fs[1][o] = (_Float16)gelu_tanh(c1[j] + b1);
    }
  }
  __syncthreads();

  // 8) x += F @ Wf2 + bf2   [16x1024]@[1024x256]
  for (int nt = wave; nt < 16; nt += 4) {
    v8f c0 = v8f_zero(), c1 = v8f_zero();
    wmma_tile2<1024>(Fs[0], Fs[1], 1040, W.Wf2T + nt * 16 * 1024, 1024, lane,
                     c0, c1);
#pragma unroll
    for (int j = 0; j < 8; ++j) {
      int n = nt * 16 + r;
      int o = (j + 8 * g) * 256 + n;
      float b2 = W.bf2[n];
      Xs[0][o] += c0[j] + b2;
      Xs[1][o] += c1[j] + b2;
    }
  }
  __syncthreads();

  // 9) cls = relu(x) @ clsW + clsB  (row 0 of each M-tile emitted)
  for (int idx = tid; idx < RPB * 16 * 256; idx += 128) {
    int rr = idx >> 12, li = idx & 4095;
    Hs[rr][(li >> 8) * 264 + (li & 255)] = (_Float16)fmaxf(Xs[rr][li], 0.f);
  }
  __syncthreads();
  for (int nt = wave; nt < 16; nt += 4) {
    v8f c0 = v8f_zero(), c1 = v8f_zero();
    wmma_tile2<256>(Hs[0], Hs[1], 264, W.clsWT + nt * 16 * 256, 256, lane,
                    c0, c1);
    if (g == 0) {  // c[0] with g==0 is M=0
      int n = nt * 16 + r;
      float cb = W.clsB[n];
      clsOut[(size_t)(row0 + 0) * 256 + n] = c0[0] + cb;
      clsOut[(size_t)(row0 + 1) * 256 + n] = c1[0] + cb;
    }
  }
  __syncthreads();

  // 10) tok = relu(ln(x;tl)) @ tlW + tlB -> next-layer slots (shift +1)
  ln_rows(Xs[0], Hs[0], W.tlg, W.tlb, red, tid, true);
  __syncthreads();
  ln_rows(Xs[1], Hs[1], W.tlg, W.tlb, red, tid, true);
  __syncthreads();
  for (int nt = wave; nt < 16; nt += 4) {
    v8f c0 = v8f_zero(), c1 = v8f_zero();
    wmma_tile2<256>(Hs[0], Hs[1], 264, W.tlWT + nt * 16 * 256, 256, lane,
                    c0, c1);
#pragma unroll
    for (int j = 0; j < 8; ++j) {
      int m = j + 8 * g, n = nt * 16 + r;
      if (m < s) {
        float tb = W.tlB[n];
        tokOut[(size_t)(row0 + 0) * TOK_STRIDE + (size_t)(m + 1) * 256 + n] =
            (_Float16)(c0[j] + tb);
        tokOut[(size_t)(row0 + 1) * TOK_STRIDE + (size_t)(m + 1) * 256 + n] =
            (_Float16)(c1[j] + tb);
      }
    }
  }

  // 11) zero pad slots (0 and s+1..15) of output buffers
  for (int idx = tid; idx < RPB * 16 * 256; idx += 128) {
    int rr = idx >> 12, li = idx & 4095;
    int slot = li >> 8;
    if (slot == 0 || slot > s)
      tokOut[(size_t)(row0 + rr) * TOK_STRIDE + li] = (_Float16)0.f;
  }
}

// ---------------------------------------------------------------------------
// FK gather of source CLS vectors, add into orders token slots 13 ("-1") and
// 12 ("-2") of the layer-1 output (13 tokens occupy slots 1..13).
// ---------------------------------------------------------------------------
__global__ __launch_bounds__(256) void fk_add_kernel(
    _Float16* __restrict__ tokOrders, const float* __restrict__ clsU,
    const float* __restrict__ clsP, const int* __restrict__ fku,
    const int* __restrict__ fkp) {
  const int row = blockIdx.x, d = threadIdx.x;
  const size_t base = (size_t)row * TOK_STRIDE;
  const int u = fku[row], p = fkp[row];
  _Float16* a = tokOrders + base + 13 * DIM + d;
  *a = (_Float16)((float)*a + clsU[(size_t)u * DIM + d]);
  _Float16* b = tokOrders + base + 12 * DIM + d;
  *b = (_Float16)((float)*b + clsP[(size_t)p * DIM + d]);
}

// ---------------------------------------------------------------------------
// Final head: out = relu(ln(cls_orders)) @ W + b, 16 rows per wave32 block.
// ---------------------------------------------------------------------------
__global__ __launch_bounds__(32) void head_kernel(
    const float* __restrict__ clsO, const _Float16* __restrict__ outWT,
    const float* __restrict__ lng, const float* __restrict__ lnb,
    const float* __restrict__ outB, float* __restrict__ out) {
  __shared__ _Float16 A[16 * 264];
  const int lane = threadIdx.x;
  const int base = blockIdx.x * 16;
  if (lane < 16) {
    const float* xr = clsO + (size_t)(base + lane) * 256;
    float s0 = 0.f, s1 = 0.f;
    for (int i = 0; i < 256; ++i) { float v = xr[i]; s0 += v; s1 += v * v; }
    float m = s0 * (1.f / 256.f);
    float q = s1 * (1.f / 256.f) - m * m;
    float rs = rsqrtf(q + 1e-5f);
    for (int i = 0; i < 256; ++i) {
      float v = (xr[i] - m) * rs * lng[i] + lnb[i];
      A[lane * 264 + i] = (_Float16)fmaxf(v, 0.f);
    }
  }
  __syncthreads();
  v8f c = v8f_zero();
  c = wmma_tile_acc<256>(A, 264, outWT, 256, lane, c);
  const int g = lane >> 4, r = lane & 15;
#pragma unroll
  for (int j = 0; j < 8; ++j) {
    int m = j + 8 * g;
    out[(size_t)(base + m) * DIM_OUT + r] = c[j] + outB[r];
  }
}

// ---------------------------------------------------------------------------
// Host launcher.
// Input order: setup_inputs() insertion order; "params" pytree flattened with
// JAX's sorted dict keys (ASCII: uppercase < lowercase).
//   0..8   embed.{orders,products,users}.{emb,nb,nw}
//   9..110 layers[l].{orders,products,users} x 17 leaves:
//          [Wf1,Wf2,Wo,Wqkv,bf1,bf2,cls,clsB,clsW,ln1b,ln1g,ln2b,ln2g,
//           tlB,tlW,tlb,tlg]
//   111..114 out.{W,b,lnb,lng}
//   115..122 categ_orders, numer_orders, categ_users, numer_users,
//            categ_products, numer_products, fk_orders_users, fk_orders_products
// ---------------------------------------------------------------------------
enum { W_Wf1 = 0, W_Wf2, W_Wo, W_Wqkv, W_bf1, W_bf2, W_cls, W_clsB, W_clsW,
       W_ln1b, W_ln1g, W_ln2b, W_ln2g, W_tlB, W_tlW, W_tlb, W_tlg };

static inline int layer_leaf(int l, int t, int k) { return 9 + (l * 3 + t) * 17 + k; }

extern "C" void kernel_launch(void* const* d_in, const int* in_sizes, int n_in,
                              void* d_out, int out_size, void* d_ws, size_t ws_size,
                              hipStream_t stream) {
  (void)in_sizes; (void)n_in; (void)out_size; (void)ws_size;

  auto F32 = [&](int i) { return (const float*)d_in[i]; };
  auto I32 = [&](int i) { return (const int*)d_in[i]; };

  // tables in sorted-key order: 0=orders, 1=products, 2=users
  static const int ROWS[3] = {16384, 8192, 4096};
  static const int NCAT[3] = {6, 4, 4};
  static const int NNUM[3] = {6, 4, 4};
  static const int CIDX[3] = {115, 119, 117};
  static const int NIDX[3] = {116, 120, 118};
  const int IDX_OUT_W = 111, IDX_OUT_b = 112, IDX_OUT_lnb = 113, IDX_OUT_lng = 114;
  const int IDX_FK_U = 121, IDX_FK_P = 122;

  // ---- workspace carve (256B aligned bump allocator) ----
  uintptr_t wsbase = (uintptr_t)d_ws;
  size_t off = 0;
  auto alloc = [&](size_t bytes) -> void* {
    off = (off + 255) & ~(size_t)255;
    void* p = (void*)(wsbase + off);
    off += bytes;
    return p;
  };

  _Float16 *wqkvT[2][3], *woT[2][3], *wf1T[2][3], *wf2T[2][3],
           *clswT[2][3], *tlwT[2][3];
  for (int l = 0; l < 2; ++l)
    for (int t = 0; t < 3; ++t) {
      wqkvT[l][t] = (_Float16*)alloc((size_t)384 * 256 * 2);
      woT[l][t]   = (_Float16*)alloc((size_t)256 * 128 * 2);
      wf1T[l][t]  = (_Float16*)alloc((size_t)1024 * 256 * 2);
      wf2T[l][t]  = (_Float16*)alloc((size_t)256 * 1024 * 2);
      clswT[l][t] = (_Float16*)alloc((size_t)256 * 256 * 2);
      tlwT[l][t]  = (_Float16*)alloc((size_t)256 * 256 * 2);
    }
  _Float16* outWT = (_Float16*)alloc((size_t)16 * 256 * 2);

  _Float16 *tokA[3], *tokB[3];
  float* clsBuf[3];
  for (int t = 0; t < 3; ++t) {
    tokA[t]   = (_Float16*)alloc((size_t)ROWS[t] * TOK_STRIDE * 2);
    tokB[t]   = (_Float16*)alloc((size_t)ROWS[t] * TOK_STRIDE * 2);
    clsBuf[t] = (float*)alloc((size_t)ROWS[t] * DIM * 4);
  }

  // ---- 1) weight convert+transpose to f16 ----
  auto xp = [&](const float* src, _Float16* dst, int K, int N) {
    int total = K * N;
    xpose_f32_to_f16<<<(total + 255) / 256, 256, 0, stream>>>(src, dst, K, N);
  };
  for (int l = 0; l < 2; ++l)
    for (int t = 0; t < 3; ++t) {
      xp(F32(layer_leaf(l, t, W_Wqkv)), wqkvT[l][t], 256, 384);
      xp(F32(layer_leaf(l, t, W_Wo)),   woT[l][t],   128, 256);
      xp(F32(layer_leaf(l, t, W_Wf1)),  wf1T[l][t],  256, 1024);
      xp(F32(layer_leaf(l, t, W_Wf2)),  wf2T[l][t],  1024, 256);
      xp(F32(layer_leaf(l, t, W_clsW)), clswT[l][t], 256, 256);
      xp(F32(layer_leaf(l, t, W_tlW)),  tlwT[l][t],  256, 256);
    }
  xp(F32(IDX_OUT_W), outWT, 256, 16);

  // ---- 2) embeddings -> tokA ----
  for (int t = 0; t < 3; ++t) {
    embed_kernel<<<ROWS[t], 256, 0, stream>>>(
        I32(CIDX[t]), F32(NIDX[t]),
        F32(t * 3 + 0) /*emb*/, F32(t * 3 + 2) /*nw*/, F32(t * 3 + 1) /*nb*/,
        tokA[t], NCAT[t], NNUM[t]);
  }

  // ---- 3) layers ----
  for (int l = 0; l < 2; ++l) {
    for (int t = 0; t < 3; ++t) {
      LayerW w;
      w.WqkvT = wqkvT[l][t]; w.WoT = woT[l][t];
      w.Wf1T = wf1T[l][t];   w.Wf2T = wf2T[l][t];
      w.clsWT = clswT[l][t]; w.tlWT = tlwT[l][t];
      w.clsTok = F32(layer_leaf(l, t, W_cls));
      w.ln1g = F32(layer_leaf(l, t, W_ln1g)); w.ln1b = F32(layer_leaf(l, t, W_ln1b));
      w.ln2g = F32(layer_leaf(l, t, W_ln2g)); w.ln2b = F32(layer_leaf(l, t, W_ln2b));
      w.bf1  = F32(layer_leaf(l, t, W_bf1));  w.bf2  = F32(layer_leaf(l, t, W_bf2));
      w.clsB = F32(layer_leaf(l, t, W_clsB));
      w.tlg  = F32(layer_leaf(l, t, W_tlg));  w.tlb  = F32(layer_leaf(l, t, W_tlb));
      w.tlB  = F32(layer_leaf(l, t, W_tlB));
      const int s = 1 + NCAT[t] + NNUM[t] + l;  // seq incl CLS, grows per layer
      const _Float16* in = (l == 0) ? tokA[t] : tokB[t];
      _Float16* out      = (l == 0) ? tokB[t] : tokA[t];
      table_block_kernel<<<ROWS[t] / RPB, 128, 0, stream>>>(in, out, clsBuf[t],
                                                            w, s);
    }
    if (l == 0) {
      // FK scatter-add into orders layer-1 token output (layer-2 FK is unused
      // by the final head, so it is skipped).
      fk_add_kernel<<<ROWS[0], 256, 0, stream>>>(
          tokB[0], clsBuf[2] /*users*/, clsBuf[1] /*products*/,
          I32(IDX_FK_U), I32(IDX_FK_P));
    }
  }

  // ---- 4) output head on orders CLS (layer 2) ----
  head_kernel<<<ROWS[0] / 16, 32, 0, stream>>>(
      clsBuf[0], outWT, F32(IDX_OUT_lng), F32(IDX_OUT_lnb), F32(IDX_OUT_b),
      (float*)d_out);
}
